// GATv2_70643622084725
// MI455X (gfx1250) — compile-verified, matrix operand order checked by reading, not run on hardware
//
#include <hip/hip_runtime.h>
#include <hip/hip_bf16.h>

// ---------- types ----------
typedef __attribute__((ext_vector_type(16))) __bf16 v16bf;
typedef __attribute__((ext_vector_type(8)))  __bf16 v8bf;
typedef __attribute__((ext_vector_type(8)))  float  v8f;
typedef int a4i __attribute__((vector_size(16)));   // matches async-lds builtin param type

#define CDIV(a, b) (((a) + (b) - 1) / (b))

// ---------- helpers ----------
static __device__ __forceinline__ __bf16 f2bf(float f) {
  unsigned u = __float_as_uint(f);
  unsigned r = 0x7FFFu + ((u >> 16) & 1u);          // round-to-nearest-even
  unsigned short h = (unsigned short)((u + r) >> 16);
  return __builtin_bit_cast(__bf16, h);
}

// order-preserving float <-> uint key (for atomic max on floats of any sign)
static __device__ __forceinline__ unsigned fkey(float f) {
  unsigned u = __float_as_uint(f);
  return (u & 0x80000000u) ? ~u : (u | 0x80000000u);
}
static __device__ __forceinline__ float keyf(unsigned k) {
  unsigned u = (k & 0x80000000u) ? (k ^ 0x80000000u) : ~k;
  return __uint_as_float(u);
}

// ---------- fp32 -> bf16 bulk convert ----------
__global__ void cvt_f32_bf16(const float* __restrict__ in, __bf16* __restrict__ out, int n) {
  int i = blockIdx.x * blockDim.x + threadIdx.x;
  if (i < n) out[i] = f2bf(in[i]);
}

// ---------- pack W (din x 256, row-major fp32) into per-lane B fragments ----------
// Bp layout: [kt][nt][lane][16 bf16]; lane L covers col = nt*16 + (L&15),
// K = kt*32 + (L<16 ? {0..7,16..23} : {8..15,24..31})
__global__ void pack_w(const float* __restrict__ W, __bf16* __restrict__ Bp, int K) {
  int idx = blockIdx.x * blockDim.x + threadIdx.x;
  int total = (K >> 5) * 16 * 32;
  if (idx >= total) return;
  int lane = idx & 31;
  int nt   = (idx >> 5) & 15;
  int kt   = idx >> 9;
  int col  = nt * 16 + (lane & 15);
  int k0   = kt * 32 + ((lane < 16) ? 0 : 8);
  __bf16* dst = Bp + (size_t)idx * 16;
#pragma unroll
  for (int i = 0; i < 16; ++i) {
    int k = k0 + ((i < 8) ? i : (8 + i));           // {k0..k0+7, k0+16..k0+23}
    dst[i] = f2bf(W[(size_t)k * 256 + col]);
  }
}

// ---------- WMMA GEMM: C[N x 256] = A[N x K](bf16) * W(packed bf16), fp32 acc ----------
// Block = 4 waves, each wave one 16-row M tile; block covers 4 column tiles (64 cols).
// Packed-B slab for the block (all K, 4 col tiles, max 16 KB) is staged to LDS once
// via async global->LDS copies (CDNA5 ASYNCcnt path), then WMMA feeds from ds loads.
__global__ __launch_bounds__(128) void gemm_wmma_bf16(
    const __bf16* __restrict__ A, const __bf16* __restrict__ Bp,
    float* __restrict__ C, int Mtiles, int K) {
  const int ldc = 256;
  __shared__ __attribute__((aligned(32))) __bf16 Bs[4 * 4 * 512];  // [kt][nt][lane][16], K<=128

  int wave = threadIdx.x >> 5;
  int lane = threadIdx.x & 31;
  int mt = blockIdx.x * 4 + wave;
  int ntBase = blockIdx.y * 4;                       // 4 col tiles of 16 -> 64 cols
  int ktiles = K >> 5;

  // ---- stage packed B for this block's 4 column tiles into LDS (all waves) ----
  int chunks = ktiles * 256;                         // 16B chunks: ktiles * 4096B / 16
#if __has_builtin(__builtin_amdgcn_global_load_async_to_lds_b128)
  for (int c = threadIdx.x; c < chunks; c += 128) {
    int kt = c >> 8, rem = c & 255;
    const __bf16* g = Bp + ((size_t)(kt * 16 + ntBase)) * 512 + rem * 8;
    __builtin_amdgcn_global_load_async_to_lds_b128(
        (__attribute__((address_space(1))) a4i*)(unsigned long long)(size_t)g,
        (__attribute__((address_space(3))) a4i*)(unsigned int)(size_t)(Bs + (size_t)c * 8),
        0, 0);
  }
  asm volatile("s_wait_asynccnt 0x0" ::: "memory");
#else
  for (int c = threadIdx.x; c < chunks; c += 128) {
    int kt = c >> 8, rem = c & 255;
    const v8bf* g = (const v8bf*)(Bp + ((size_t)(kt * 16 + ntBase)) * 512 + rem * 8);
    *(v8bf*)(Bs + (size_t)c * 8) = *g;
  }
#endif
  __syncthreads();

  if (mt < Mtiles) {                                 // wave-uniform guard: EXEC all-1 for WMMA
    int row  = lane & 15;
    int koff = (lane < 16) ? 0 : 8;
    const __bf16* arow = A + (size_t)(mt * 16 + row) * K;

    v8f acc0 = {}, acc1 = {}, acc2 = {}, acc3 = {};
    for (int kt = 0; kt < ktiles; ++kt) {
      int kb = kt * 32 + koff;
      __builtin_prefetch(arow + kb + 32, 0, 3);      // global_prefetch_b8 (gfx1250)
      v8bf a0 = *(const v8bf*)(arow + kb);
      v8bf a1 = *(const v8bf*)(arow + kb + 16);
      v16bf av;
#pragma unroll
      for (int i = 0; i < 8; ++i) { av[i] = a0[i]; av[8 + i] = a1[i]; }

      const __bf16* bbase = Bs + (size_t)kt * 2048 + lane * 16;
      v16bf b0 = *(const v16bf*)(bbase);
      v16bf b1 = *(const v16bf*)(bbase + 512);
      v16bf b2 = *(const v16bf*)(bbase + 1024);
      v16bf b3 = *(const v16bf*)(bbase + 1536);

      acc0 = __builtin_amdgcn_wmma_f32_16x16x32_bf16(false, av, false, b0, (short)0, acc0, false, false);
      acc1 = __builtin_amdgcn_wmma_f32_16x16x32_bf16(false, av, false, b1, (short)0, acc1, false, false);
      acc2 = __builtin_amdgcn_wmma_f32_16x16x32_bf16(false, av, false, b2, (short)0, acc2, false, false);
      acc3 = __builtin_amdgcn_wmma_f32_16x16x32_bf16(false, av, false, b3, (short)0, acc3, false, false);
    }

    // C/D layout: vgpr r -> M = r (lanes 0-15) / r+8 (lanes 16-31); N = lane%16
    int n  = lane & 15;
    int mb = (lane < 16) ? 0 : 8;
    float* cp = C + (size_t)(mt * 16 + mb) * ldc + ntBase * 16 + n;
#pragma unroll
    for (int r = 0; r < 8; ++r) {
      cp[(size_t)r * ldc +  0] = acc0[r];
      cp[(size_t)r * ldc + 16] = acc1[r];
      cp[(size_t)r * ldc + 32] = acc2[r];
      cp[(size_t)r * ldc + 48] = acc3[r];
    }
  }
}

// ---------- edge logits + segment max (one thread per (edge, head)) ----------
__global__ void edge_logits(const float* __restrict__ xl, const float* __restrict__ xr,
                            const int* __restrict__ src, const int* __restrict__ dst,
                            const float* __restrict__ att, float* __restrict__ logits,
                            unsigned* __restrict__ mkey, int E) {
  int idx = blockIdx.x * blockDim.x + threadIdx.x;
  if (idx >= E * 4) return;
  int h = idx & 3, e = idx >> 2;
  int s = src[e], d = dst[e];
  const float4* pl = (const float4*)(xl + (size_t)s * 256 + h * 64);
  const float4* pr = (const float4*)(xr + (size_t)d * 256 + h * 64);
  const float4* pa = (const float4*)(att + h * 64);
  float acc = 0.f;
#pragma unroll 4
  for (int i = 0; i < 16; ++i) {
    float4 a = pl[i], b = pr[i], w = pa[i];
    float ex = a.x + b.x, ey = a.y + b.y, ez = a.z + b.z, ew = a.w + b.w;
    ex = ex > 0.f ? ex : 0.2f * ex;  ey = ey > 0.f ? ey : 0.2f * ey;
    ez = ez > 0.f ? ez : 0.2f * ez;  ew = ew > 0.f ? ew : 0.2f * ew;
    acc += ex * w.x + ey * w.y + ez * w.z + ew * w.w;
  }
  logits[idx] = acc;
  atomicMax(mkey + (size_t)d * 4 + h, fkey(acc));
}

// ---------- exp(logit - max) + segment denom ----------
__global__ void edge_softmax_p(float* __restrict__ logits, const unsigned* __restrict__ mkey,
                               float* __restrict__ denom, const int* __restrict__ dst, int E) {
  int idx = blockIdx.x * blockDim.x + threadIdx.x;
  if (idx >= E * 4) return;
  int h = idx & 3, e = idx >> 2;
  int d = dst[e];
  float m = keyf(mkey[(size_t)d * 4 + h]);
  float p = __expf(logits[idx] - m);
  logits[idx] = p;                                   // reuse buffer for p
  atomicAdd(denom + (size_t)d * 4 + h, p);
}

// ---------- scatter: out[dst] += alpha * xl[src] ----------
__global__ void edge_scatter(const float* __restrict__ p, const float* __restrict__ denom,
                             const float* __restrict__ xl, const int* __restrict__ src,
                             const int* __restrict__ dst, float* __restrict__ out, int E) {
  int idx = blockIdx.x * blockDim.x + threadIdx.x;
  if (idx >= E * 4) return;
  int h = idx & 3, e = idx >> 2;
  int s = src[e], d = dst[e];
  float alpha = p[idx] / (denom[(size_t)d * 4 + h] + 1e-16f);
  const float4* pl = (const float4*)(xl + (size_t)s * 256 + h * 64);
  float* po = out + (size_t)d * 256 + h * 64;
#pragma unroll 4
  for (int i = 0; i < 16; ++i) {
    float4 a = pl[i];
    atomicAdd(po + 4 * i + 0, alpha * a.x);
    atomicAdd(po + 4 * i + 1, alpha * a.y);
    atomicAdd(po + 4 * i + 2, alpha * a.z);
    atomicAdd(po + 4 * i + 3, alpha * a.w);
  }
}

// ---------- head mean + bias + leaky(0.1); emit fp32 h and bf16 h ----------
__global__ void node_finish(const float* __restrict__ acc, const float* __restrict__ bias,
                            float* __restrict__ hfp, __bf16* __restrict__ hbf, int N) {
  int idx = blockIdx.x * blockDim.x + threadIdx.x;
  if (idx >= N * 64) return;
  int n = idx >> 6, d = idx & 63;
  const float* a = acc + (size_t)n * 256 + d;
  float v = 0.25f * (a[0] + a[64] + a[128] + a[192]) + bias[d];
  v = v > 0.f ? v : 0.1f * v;
  hfp[idx] = v;
  hbf[idx] = f2bf(v);
}

// ---------- graph pooling ----------
__global__ void pool_sum(const float* __restrict__ h, const int* __restrict__ batch,
                         float* __restrict__ pooled, int N) {
  int idx = blockIdx.x * blockDim.x + threadIdx.x;
  if (idx >= N * 64) return;
  int n = idx >> 6, d = idx & 63;
  atomicAdd(pooled + (size_t)batch[n] * 64 + d, h[idx]);
}

// ---------- BN over graphs + FC head (single block) ----------
__global__ __launch_bounds__(256) void final_head(
    const float* __restrict__ pooled, const float* __restrict__ gamma,
    const float* __restrict__ beta, const float* __restrict__ fcW,
    const float* __restrict__ fcb, float* __restrict__ out) {
  __shared__ float norm[64 * 64];
  int t = threadIdx.x;
  if (t < 64) {
    float s = 0.f, s2 = 0.f;
    for (int g = 0; g < 64; ++g) { float v = pooled[g * 64 + t]; s += v; s2 += v * v; }
    float mean = s * (1.f / 64.f);
    float var  = s2 * (1.f / 64.f) - mean * mean;     // biased variance
    float inv  = rsqrtf(var + 1e-5f) * gamma[t];
    float bt   = beta[t];
    for (int g = 0; g < 64; ++g) norm[g * 64 + t] = (pooled[g * 64 + t] - mean) * inv + bt;
  }
  __syncthreads();
  for (int i = t; i < 64 * 32; i += 256) {
    int g = i >> 5, l = i & 31;
    float acc = fcb[l];
    const float* ng = norm + g * 64;
    const float* wl = fcW + l * 64;
#pragma unroll 8
    for (int d = 0; d < 64; ++d) acc += ng[d] * wl[d];
    out[i] = acc;
  }
}

// ---------- host side ----------
extern "C" void kernel_launch(void* const* d_in, const int* in_sizes, int n_in,
                              void* d_out, int out_size, void* d_ws, size_t ws_size,
                              hipStream_t stream) {
  (void)n_in; (void)out_size; (void)ws_size;
  const float* x      = (const float*)d_in[0];
  const int*   eidx   = (const int*)d_in[1];
  const int*   batch  = (const int*)d_in[2];
  const float* Wl[3]  = {(const float*)d_in[3],  (const float*)d_in[7],  (const float*)d_in[11]};
  const float* Wr[3]  = {(const float*)d_in[4],  (const float*)d_in[8],  (const float*)d_in[12]};
  const float* att[3] = {(const float*)d_in[5],  (const float*)d_in[9],  (const float*)d_in[13]};
  const float* bia[3] = {(const float*)d_in[6],  (const float*)d_in[10], (const float*)d_in[14]};
  const float* gamma  = (const float*)d_in[15];
  const float* beta   = (const float*)d_in[16];
  const float* fcW    = (const float*)d_in[17];
  const float* fcb    = (const float*)d_in[18];
  float* out          = (float*)d_out;

  const int N = in_sizes[0] / 128;
  const int E = in_sizes[1] / 2;
  const int* src = eidx;
  const int* dst = eidx + E;
  const int Mtiles = N / 16;

  // workspace carving (256B aligned)
  char* ws = (char*)d_ws;
  size_t off = 0;
  auto take = [&](size_t bytes) -> char* {
    char* p = ws + off;
    off += (bytes + 255) & ~(size_t)255;
    return p;
  };
  __bf16* xbf    = (__bf16*)take((size_t)N * 128 * 2);
  __bf16* hbf    = (__bf16*)take((size_t)N * 64 * 2);
  float*  xl     = (float*)take((size_t)N * 256 * 4);
  float*  xr     = (float*)take((size_t)N * 256 * 4);
  float*  outacc = (float*)take((size_t)N * 256 * 4);
  float*  logits = (float*)take((size_t)E * 4 * 4);
  unsigned* mkey = (unsigned*)take((size_t)N * 4 * 4);
  float*  denom  = (float*)take((size_t)N * 4 * 4);
  float*  hfp    = (float*)take((size_t)N * 64 * 4);
  float*  pooled = (float*)take(64 * 64 * 4);
  __bf16* pWl[3], *pWr[3];
  const int din[3] = {128, 64, 64};
  for (int l = 0; l < 3; ++l) {
    pWl[l] = (__bf16*)take((size_t)din[l] * 256 * 2);
    pWr[l] = (__bf16*)take((size_t)din[l] * 256 * 2);
  }

  // x -> bf16
  cvt_f32_bf16<<<CDIV(N * 128, 256), 256, 0, stream>>>(x, xbf, N * 128);

  const __bf16* abf = xbf;
  for (int l = 0; l < 3; ++l) {
    int K = din[l];
    int packThreads = (K >> 5) * 16 * 32;
    pack_w<<<CDIV(packThreads, 256), 256, 0, stream>>>(Wl[l], pWl[l], K);
    pack_w<<<CDIV(packThreads, 256), 256, 0, stream>>>(Wr[l], pWr[l], K);

    dim3 gg(CDIV(Mtiles, 4), 4);
    gemm_wmma_bf16<<<gg, 128, 0, stream>>>(abf, pWl[l], xl, Mtiles, K);
    gemm_wmma_bf16<<<gg, 128, 0, stream>>>(abf, pWr[l], xr, Mtiles, K);

    (void)hipMemsetAsync(outacc, 0, (size_t)N * 256 * 4, stream);
    (void)hipMemsetAsync(mkey,   0, (size_t)N * 4 * 4,   stream);
    (void)hipMemsetAsync(denom,  0, (size_t)N * 4 * 4,   stream);

    int EH = E * 4;
    edge_logits   <<<CDIV(EH, 256), 256, 0, stream>>>(xl, xr, src, dst, att[l], logits, mkey, E);
    edge_softmax_p<<<CDIV(EH, 256), 256, 0, stream>>>(logits, mkey, denom, dst, E);
    edge_scatter  <<<CDIV(EH, 256), 256, 0, stream>>>(logits, denom, xl, src, dst, outacc, E);
    node_finish   <<<CDIV(N * 64, 256), 256, 0, stream>>>(outacc, bia[l], hfp, hbf, N);
    abf = hbf;   // next layer consumes bf16 hidden state (K = 64)
  }

  (void)hipMemsetAsync(pooled, 0, 64 * 64 * 4, stream);
  pool_sum<<<CDIV(N * 64, 256), 256, 0, stream>>>(hfp, batch, pooled, N);
  final_head<<<1, 256, 0, stream>>>(pooled, gamma, beta, fcW, fcb, out);
}
